// MultiHeadAttention_33389075759360
// MI455X (gfx1250) — compile-verified, matrix operand order checked by reading, not run on hardware
//
#include <hip/hip_runtime.h>

typedef unsigned short u16t;
typedef __attribute__((ext_vector_type(16))) __bf16        v16bf;
typedef __attribute__((ext_vector_type(8)))  float         v8f;
typedef __attribute__((ext_vector_type(4)))  unsigned int  v4u;
typedef __attribute__((ext_vector_type(2)))  unsigned int  v2u;
typedef __attribute__((ext_vector_type(4)))  float         v4f;

#define SEQ   2048
#define DM    1024
#define HEADS 16
#define DK    64

union ABu { v16bf v; v4u q[2]; };

static __device__ __forceinline__ unsigned short f2bf(float f) {
  union { float f; unsigned int u; } x;
  x.f = f;
  unsigned int r = x.u + 0x7FFFu + ((x.u >> 16) & 1u);
  return (unsigned short)(r >> 16);
}

static __device__ __forceinline__ v8f wmma_bf16(v16bf a, v16bf b, v8f c) {
  return __builtin_amdgcn_wmma_f32_16x16x32_bf16(false, a, false, b, (short)0, c,
                                                 false, false);
}

// WMMA A/B operand from a row-major bf16 matrix.
// Lanes 0-15 hold K {0..7,16..23}; lanes 16-31 hold K {8..15,24..31}.
static __device__ __forceinline__ v16bf load_tile(const u16t* __restrict__ base,
                                                  int row, int ld, int k0, int lane) {
  const int klo = (lane & 16) ? 8 : 0;
  const u16t* p = base + (size_t)row * (size_t)ld + (size_t)(k0 + klo);
  ABu ab;
  ab.q[0] = *(const v4u*)(p);
  ab.q[1] = *(const v4u*)(p + 16);
  return ab.v;
}

// Same operand layout from an LDS tile row slice (16B-aligned chunks).
static __device__ __forceinline__ v16bf load_tile_lds(const u16t* p0, int lane) {
  const int klo = (lane & 16) ? 8 : 0;
  ABu ab;
  ab.q[0] = *(const v4u*)(p0 + klo);
  ab.q[1] = *(const v4u*)(p0 + klo + 16);
  return ab.v;
}

// CDNA5 async copy: 16 bytes/lane global -> LDS, tracked by ASYNCcnt.
static __device__ __forceinline__ void stage_async_b128(const u16t* g, const u16t* l) {
  unsigned lds_off = (unsigned)(uintptr_t)l;   // low 32 bits of shared aperture addr
  unsigned long long ga = (unsigned long long)(uintptr_t)g;
  asm volatile("global_load_async_to_lds_b128 %0, %1, off"
               :: "v"(lds_off), "v"(ga) : "memory");
}
static __device__ __forceinline__ void wait_async0() {
  asm volatile("s_wait_asynccnt 0x0" ::: "memory");
}

// ---------------- fp32 -> bf16 conversion (x4 vectorized) ----------------
__global__ void cvt_kernel(const float* __restrict__ s, u16t* __restrict__ d, int n) {
  int i = (blockIdx.x * blockDim.x + threadIdx.x) * 4;
  if (i < n) {
    v4f v = *(const v4f*)(s + i);
    v2u o;
    o.x = (unsigned)f2bf(v.x) | ((unsigned)f2bf(v.y) << 16);
    o.y = (unsigned)f2bf(v.z) | ((unsigned)f2bf(v.w) << 16);
    *(v2u*)(d + i) = o;
  }
}

// ================= GEMM core =================
// Block = 8 waves; each wave owns a 32x64 C-tile (2 M-tiles x 4 N-tiles).
// The block shares a double-buffered, async-staged 64(N)x64(K) W tile in LDS;
// one stage/barrier covers 16 WMMAs per wave. The K-chunk loop is manually
// unrolled x2 with ping-pong A-operand register sets (no rotate copies, no
// WMMA->VALU WAR-hazard NOPs, compile-time LDS buffer offsets).

struct GemmAcc { v8f c[2][4]; };

template <typename WriteBack>
static __device__ __forceinline__ void gemm_core(const u16t* __restrict__ A,
                                                 const u16t* __restrict__ W,
                                                 u16t (*wt)[64][64],   // [2][64][64]
                                                 WriteBack&& wb) {
  const int tid    = threadIdx.x;
  const int lane   = tid & 31;
  const int wv     = tid >> 5;
  const int nb     = blockIdx.x & 15;             // N block shared by all waves
  const int mtp    = (blockIdx.x >> 4) * 8 + wv;  // per-wave 32-row group
  const int mbase  = mtp * 32;
  const int n0     = nb * 64;
  const int col    = lane & 15;

  // async-stage mapping: 256 threads x 32B (2 x b128) cover a 64x64 bf16 tile
  const int srow = tid >> 2, schunk = tid & 3;
  const u16t* gstage = W + (size_t)(n0 + srow) * DM + schunk * 8;
  u16t* l0 = &wt[0][srow][schunk * 8];
  u16t* l1 = &wt[1][srow][schunk * 8];

  GemmAcc acc;
  v8f z = {};
#pragma unroll
  for (int i = 0; i < 2; ++i)
#pragma unroll
    for (int t = 0; t < 4; ++t) acc.c[i][t] = z;

  const int arow0 = mbase + col;
  const int arow1 = mbase + 16 + col;

  auto load_a4 = [&](v16bf a[2][2], int k0) {
    a[0][0] = load_tile(A, arow0, DM, k0,      lane);
    a[0][1] = load_tile(A, arow1, DM, k0,      lane);
    a[1][0] = load_tile(A, arow0, DM, k0 + 32, lane);
    a[1][1] = load_tile(A, arow1, DM, k0 + 32, lane);
  };
  auto compute = [&](const u16t (*buf)[64], const v16bf a[2][2]) {
#pragma unroll
    for (int s = 0; s < 2; ++s) {
      v16bf bw[4];
#pragma unroll
      for (int t = 0; t < 4; ++t)                 // one clause, graduated waits
        bw[t] = load_tile_lds(&buf[t * 16 + col][s * 32], lane);
#pragma unroll
      for (int t = 0; t < 4; ++t) {
        acc.c[0][t] = wmma_bf16(a[s][0], bw[t], acc.c[0][t]);
        acc.c[1][t] = wmma_bf16(a[s][1], bw[t], acc.c[1][t]);
      }
    }
  };

  v16bf aqA[2][2], aqB[2][2];

  // prologue: stage chunk 0 -> buf0, load its A operands
  stage_async_b128(gstage,      l0);
  stage_async_b128(gstage + 32, l0 + 32);
  load_a4(aqA, 0);
  wait_async0();
  __syncthreads();

  for (int kc = 0; kc < DM; kc += 128) {
    {   // half A: compute chunk kc from buf0; prefetch kc+64 -> buf1
      const int nk = kc + 64;                     // always < DM
      stage_async_b128(gstage + nk,      l1);
      stage_async_b128(gstage + nk + 32, l1 + 32);
      load_a4(aqB, nk);
      compute(wt[0], aqA);
      wait_async0();
      __syncthreads();
    }
    {   // half B: compute chunk kc+64 from buf1; prefetch kc+128 -> buf0
      const int nk = kc + 128;
      if (nk < DM) {
        stage_async_b128(gstage + nk,      l0);
        stage_async_b128(gstage + nk + 32, l0 + 32);
        load_a4(aqA, nk);
      }
      compute(wt[1], aqB);
      wait_async0();
      __syncthreads();
    }
  }

  wb(acc, mbase, n0);
}

// mode 0: write bf16 [b,h,s,dk] (Q, K); mode 2: write bf16 [b,h,dk,s] (V^T)
__global__ void __launch_bounds__(256) proj_kernel(const u16t* __restrict__ A,
                                                   const u16t* __restrict__ W,
                                                   u16t* __restrict__ out, int mode) {
  __shared__ __align__(16) u16t wt[2][64][64];
  const int lane   = threadIdx.x & 31;
  const int col    = lane & 15;
  const int rowoff = (lane & 16) ? 8 : 0;
  gemm_core(A, W, wt, [&](const GemmAcc& acc, int mbase, int n0) {
#pragma unroll
    for (int i = 0; i < 2; ++i) {
#pragma unroll
      for (int t = 0; t < 4; ++t) {
        const int n = n0 + t * 16 + col;
        const int h = n >> 6, d = n & 63;
#pragma unroll
        for (int r = 0; r < 8; ++r) {
          const int m = mbase + i * 16 + r + rowoff;
          const int bb = m >> 11, s = m & 2047;
          size_t idx;
          if (mode == 2) idx = (((size_t)(bb * HEADS + h)) * DK + d) * SEQ + s;
          else           idx = (((size_t)(bb * HEADS + h)) * SEQ + s) * DK + d;
          out[idx] = f2bf(acc.c[i][t][r]);
        }
      }
    }
  });
}

__global__ void __launch_bounds__(256) out_kernel(const u16t* __restrict__ A,
                                                  const u16t* __restrict__ W,
                                                  const float* __restrict__ bias,
                                                  float* __restrict__ out) {
  __shared__ __align__(16) u16t wt[2][64][64];
  const int lane   = threadIdx.x & 31;
  const int col    = lane & 15;
  const int rowoff = (lane & 16) ? 8 : 0;
  gemm_core(A, W, wt, [&](const GemmAcc& acc, int mbase, int n0) {
#pragma unroll
    for (int i = 0; i < 2; ++i) {
#pragma unroll
      for (int t = 0; t < 4; ++t) {
        const int n = n0 + t * 16 + col;
        const float bn = bias[n];
#pragma unroll
        for (int r = 0; r < 8; ++r) {
          const int m = mbase + i * 16 + r + rowoff;
          out[(size_t)m * DM + n] = acc.c[i][t][r] + bn;
        }
      }
    }
  });
}

// ---------------- streaming causal attention (flash-style) ----------------
__global__ void __launch_bounds__(256) attn_kernel(const u16t* __restrict__ Q,
                                                   const u16t* __restrict__ K,
                                                   const u16t* __restrict__ Vt,
                                                   u16t* __restrict__ C) {
  __shared__ __align__(16) u16t pbuf[8][16][40];
  const int lane   = threadIdx.x & 31;
  const int w      = threadIdx.x >> 5;
  const int wid    = blockIdx.x * 8 + w;
  const int qt     = wid & 127;
  const int h      = (wid >> 7) & 15;
  const int b      = wid >> 11;
  const int col    = lane & 15;
  const int rowoff = (lane & 16) ? 8 : 0;
  const int klo    = (lane & 16) ? 8 : 0;

  const u16t* Qh = Q  + ((size_t)(b * HEADS + h)) * SEQ * DK;
  const u16t* Kh = K  + ((size_t)(b * HEADS + h)) * SEQ * DK;
  const u16t* Vh = Vt + ((size_t)(b * HEADS + h)) * DK * SEQ;
  const int qbase = qt * 16;

  const v16bf aq0 = load_tile(Qh, qbase + col, DK, 0, lane);
  const v16bf aq1 = load_tile(Qh, qbase + col, DK, 32, lane);

  v8f z = {};
  v8f o[4];
  float rmax[8], rsum[8];
#pragma unroll
  for (int t = 0; t < 4; ++t) o[t] = z;
#pragma unroll
  for (int r = 0; r < 8; ++r) { rmax[r] = -3.0e38f; rsum[r] = 0.0f; }

  for (int kt = 0; kt <= qt; kt += 2) {
    const int kb0 = kt * 16;
    const int kb1 = kb0 + 16;      // may lie past the diagonal -> fully masked

    // batch all K operand loads (one clause/wait), then the 4 score WMMAs
    v16bf bk[4];
    bk[0] = load_tile(Kh, kb0 + col, DK, 0,  lane);
    bk[1] = load_tile(Kh, kb0 + col, DK, 32, lane);
    bk[2] = load_tile(Kh, kb1 + col, DK, 0,  lane);
    bk[3] = load_tile(Kh, kb1 + col, DK, 32, lane);
    v8f s0 = z, s1 = z;
    s0 = wmma_bf16(aq0, bk[0], s0);
    s0 = wmma_bf16(aq1, bk[1], s0);
    s1 = wmma_bf16(aq0, bk[2], s1);
    s1 = wmma_bf16(aq1, bk[3], s1);

    // V operands are score-independent: issue now so the softmax VALU/TRANS
    // work hides their latency.
    v16bf bv[4];
#pragma unroll
    for (int t = 0; t < 4; ++t)
      bv[t] = load_tile(Vh, t * 16 + col, SEQ, kb0, lane);

    if (kt + 2 <= qt)
      __builtin_prefetch(Kh + (size_t)(kb0 + 32 + col) * DK, 0, 3);

    // online softmax update (row stats replicated over the 16-lane row group)
#pragma unroll
    for (int r = 0; r < 8; ++r) {
      const int qrow = qbase + r + rowoff;
      float v0 = s0[r] * 0.125f;
      float v1 = s1[r] * 0.125f;
      if (kb0 + col > qrow) v0 = -1.0e9f;
      if (kb1 + col > qrow) v1 = -1.0e9f;
      float m = fmaxf(v0, v1);
      m = fmaxf(m, __shfl_xor(m, 1, 32));
      m = fmaxf(m, __shfl_xor(m, 2, 32));
      m = fmaxf(m, __shfl_xor(m, 4, 32));
      m = fmaxf(m, __shfl_xor(m, 8, 32));
      const float nm    = fmaxf(rmax[r], m);
      const float alpha = __expf(rmax[r] - nm);
      rmax[r] = nm;
      const float p0 = __expf(v0 - nm);
      const float p1 = __expf(v1 - nm);
      float ls = p0 + p1;
      ls += __shfl_xor(ls, 1, 32);
      ls += __shfl_xor(ls, 2, 32);
      ls += __shfl_xor(ls, 4, 32);
      ls += __shfl_xor(ls, 8, 32);
      rsum[r] = rsum[r] * alpha + ls;
#pragma unroll
      for (int t = 0; t < 4; ++t) o[t][r] *= alpha;
      pbuf[w][r + rowoff][col]      = f2bf(p0);   // C-layout -> row-major staging
      pbuf[w][r + rowoff][col + 16] = f2bf(p1);
    }

    // reload P in A-operand layout (two contiguous 16B LDS chunks per lane)
    ABu pa;
    {
      const u16t* pp = &pbuf[w][col][klo];
      pa.q[0] = *(const v4u*)(pp);
      pa.q[1] = *(const v4u*)(pp + 16);
    }

#pragma unroll
    for (int t = 0; t < 4; ++t)
      o[t] = wmma_bf16(pa.v, bv[t], o[t]);
  }

#pragma unroll
  for (int t = 0; t < 4; ++t) {
#pragma unroll
    for (int r = 0; r < 8; ++r) {
      const float val = o[t][r] / rsum[r];
      const size_t idx = ((size_t)(b * SEQ + qbase + r + rowoff)) * DM
                       + (size_t)(h * DK + t * 16 + col);
      C[idx] = f2bf(val);
    }
  }
}

extern "C" void kernel_launch(void* const* d_in, const int* in_sizes, int n_in,
                              void* d_out, int out_size, void* d_ws, size_t ws_size,
                              hipStream_t stream) {
  (void)in_sizes; (void)n_in; (void)out_size; (void)ws_size;
  const float* x  = (const float*)d_in[0];
  const float* Wq = (const float*)d_in[1];
  const float* Wk = (const float*)d_in[2];
  const float* Wv = (const float*)d_in[3];
  const float* Wo = (const float*)d_in[4];
  const float* bo = (const float*)d_in[5];
  // d_in[6] is the causal mask; regenerated analytically inside attn_kernel.

  const size_t XB  = (size_t)4096 * 1024;
  const size_t WSZ = (size_t)1024 * 1024;

  char* ws   = (char*)d_ws;
  u16t* xb   = (u16t*)ws;                         // 8 MiB (reused as concat)
  u16t* wqb  = (u16t*)(ws + 8ull * 1024 * 1024);  // 4 x 2 MiB weights
  u16t* wkb  = wqb + WSZ;
  u16t* wvb  = wkb + WSZ;
  u16t* wob  = wvb + WSZ;
  u16t* qb   = wob + WSZ;                         // 8 MiB [b,h,s,dk]
  u16t* kb   = qb + XB;                           // 8 MiB [b,h,s,dk]
  u16t* vtb  = kb + XB;                           // 8 MiB [b,h,dk,s]
  u16t* ccat = xb;                                // xb dead after projections

  cvt_kernel<<<(int)((XB / 4 + 255) / 256), 256, 0, stream>>>(x,  xb,  (int)XB);
  cvt_kernel<<<(int)((WSZ / 4 + 255) / 256), 256, 0, stream>>>(Wq, wqb, (int)WSZ);
  cvt_kernel<<<(int)((WSZ / 4 + 255) / 256), 256, 0, stream>>>(Wk, wkb, (int)WSZ);
  cvt_kernel<<<(int)((WSZ / 4 + 255) / 256), 256, 0, stream>>>(Wv, wvb, (int)WSZ);
  cvt_kernel<<<(int)((WSZ / 4 + 255) / 256), 256, 0, stream>>>(Wo, wob, (int)WSZ);

  proj_kernel<<<256, 256, 0, stream>>>(xb, wqb, qb,  0);
  proj_kernel<<<256, 256, 0, stream>>>(xb, wkb, kb,  0);
  proj_kernel<<<256, 256, 0, stream>>>(xb, wvb, vtb, 2);

  attn_kernel<<<512, 256, 0, stream>>>(qb, kb, vtb, ccat);

  out_kernel<<<256, 256, 0, stream>>>(ccat, wob, bo, (float*)d_out);
}
